// Encoder_6365141532718
// MI455X (gfx1250) — matrix operand order repaired and checked by reading, not compiled, not run on hardware
//
#include <hip/hip_runtime.h>

#define N_NODES 50000   // divisible by 16 -> every 16-row wave tile is all-in or all-out
#define N_EDGES 800000
#define HID 64
#define N_LAYERS 4

typedef __attribute__((ext_vector_type(16))) __bf16         v16bf;
typedef __attribute__((ext_vector_type(16))) unsigned short v16u;
typedef __attribute__((ext_vector_type(8)))  float          v8f;

// Native f32 -> bf16 (RNE); backend can select v_cvt_pk_bf16_f32
__device__ __forceinline__ unsigned short f2bf_bits(float f) {
    __bf16 h = (__bf16)f;
    unsigned short b;
    __builtin_memcpy(&b, &h, 2);
    return b;
}

__global__ void zero_f32(float* __restrict__ p, long n) {
    long i = (long)blockIdx.x * blockDim.x + threadIdx.x;
    if (i < n) p[i] = 0.0f;
}

__global__ void count_deg(const int* __restrict__ dst, float* __restrict__ deg) {
    int e = blockIdx.x * blockDim.x + threadIdx.x;
    if (e < N_EDGES) atomicAdd(&deg[dst[e]], 1.0f);
}

__global__ void invert_deg(float* __restrict__ deg) {
    int i = blockIdx.x * blockDim.x + threadIdx.x;
    if (i < N_NODES) deg[i] = 1.0f / fmaxf(deg[i], 1.0f);
}

// One thread per (edge, 4-channel chunk): float4 gather from x[src], 4 atomic adds into agg[dst]
__global__ void scatter_sum(const float* __restrict__ x,
                            const int* __restrict__ src, const int* __restrict__ dst,
                            float* __restrict__ agg) {
    long t = (long)blockIdx.x * blockDim.x + threadIdx.x;
    if (t >= (long)N_EDGES * 16) return;
    int e  = (int)(t >> 4);
    int c4 = (int)(t & 15);
    int s = src[e], d = dst[e];
    const float4 v = reinterpret_cast<const float4*>(x)[(long)s * 16 + c4];
    float* base = agg + (long)d * HID + c4 * 4;
    atomicAdd(base + 0, v.x);
    atomicAdd(base + 1, v.y);
    atomicAdd(base + 2, v.z);
    atomicAdd(base + 3, v.w);
}

// Fused: out = PReLU( (agg .* deg_inv) @ Wl + x @ Wr + b )
// 128 threads = 4 waves; each wave owns a 16-node tile; block covers 64 nodes.
// Weights staged in LDS pre-swizzled into WMMA B-fragment layout (32B/lane/frag,
// fetched with one aligned 32B LDS read). Tile-granular uniform bounds guard:
// since N_NODES % 16 == 0, a tile is entirely valid or entirely skipped, so the
// epilogue has zero per-element EXEC manipulation.
__global__ __launch_bounds__(128) void sage_fused_gemm(
    const float* __restrict__ agg,  const float* __restrict__ xin,
    const float* __restrict__ Wl,   const float* __restrict__ Wr,
    const float* __restrict__ bias, const float* __restrict__ alpha,
    const float* __restrict__ dinv, float* __restrict__ out) {

    // 8 fragments (4 n-tiles x 2 k-steps) x 32 lanes, 32B each = 8KB per matrix
    __shared__ v16u wl_s[8 * 32];
    __shared__ v16u wr_s[8 * 32];

    {
        unsigned short* pl = reinterpret_cast<unsigned short*>(wl_s);
        unsigned short* pr = reinterpret_cast<unsigned short*>(wr_s);
        for (int idx = threadIdx.x; idx < HID * HID; idx += 128) {
            // idx = ((nt*2+ks)*32 + lane)*16 + e  -> source element W[k][col]
            int e    = idx & 15;
            int ln   = (idx >> 4) & 31;
            int fid  = idx >> 9;         // 0..7
            int nt   = fid >> 1;
            int ks   = fid & 1;
            int hlf  = ln >> 4;
            int col  = nt * 16 + (ln & 15);
            int k    = ks * 32 + ((e >> 3) << 4) + (hlf << 3) + (e & 7);
            pl[idx] = f2bf_bits(Wl[k * HID + col]);
            pr[idx] = f2bf_bits(Wr[k * HID + col]);
        }
    }
    __syncthreads();

    const int wave = threadIdx.x >> 5;
    const int lane = threadIdx.x & 31;
    const int lrow = lane & 15;     // A: row within tile / B,C,D: column within tile
    const int half = lane >> 4;

    const int base = (blockIdx.x * 4 + wave) * 16;
    if (base < N_NODES) {   // wave-uniform: tile fully valid (N_NODES % 16 == 0)
        const int row = base + lrow;
        const float di = dinv[row];
        const float* __restrict__ arow = agg + (long)row * HID;
        const float* __restrict__ xrow = xin + (long)row * HID;

        // ---- A fragments: 16x32 bf16 layout (ISA 7.12.2), two K-steps per source ----
        v16bf a_agg[2], a_x[2];
#pragma unroll
        for (int ks = 0; ks < 2; ++ks) {
            v16bf ta, tx;
            const int kb = ks * 32;
#pragma unroll
            for (int i = 0; i < 8; ++i) {
                // VGPR i: K = kb + (i>>2)*16 + half*8 + (i&3)*2 , +1
                int k = kb + ((i >> 2) << 4) + (half << 3) + ((i & 3) << 1);
                ta[2*i]   = (__bf16)(arow[k]     * di);
                ta[2*i+1] = (__bf16)(arow[k + 1] * di);
                tx[2*i]   = (__bf16)(xrow[k]);
                tx[2*i+1] = (__bf16)(xrow[k + 1]);
            }
            a_agg[ks] = ta;
            a_x[ks]   = tx;
        }

        // Hoist epilogue constants so no loads land between WMMAs
        float bc[4], al[4];
#pragma unroll
        for (int nt = 0; nt < 4; ++nt) {
            bc[nt] = bias[nt * 16 + lrow];
            al[nt] = alpha[nt * 16 + lrow];
        }

        // Common store base: rows base+half*8 .. +7, immediate offsets per j
        float* __restrict__ op = out + (long)(base + (half << 3)) * HID;

#pragma unroll
        for (int nt = 0; nt < 4; ++nt) {
            const int col = nt * 16 + lrow;
            v8f acc = {};

            // agg @ Wl : two 16x16x32 WMMAs; B fragment = one 32B LDS read
#pragma unroll
            for (int ks = 0; ks < 2; ++ks) {
                v16bf bf = __builtin_bit_cast(v16bf, wl_s[(nt * 2 + ks) * 32 + lane]);
                acc = __builtin_amdgcn_wmma_f32_16x16x32_bf16(
                    false, a_agg[ks], false, bf, (short)0, acc, false, false);
            }
            // x @ Wr : two more WMMAs into same accumulator
#pragma unroll
            for (int ks = 0; ks < 2; ++ks) {
                v16bf bf = __builtin_bit_cast(v16bf, wr_s[(nt * 2 + ks) * 32 + lane]);
                acc = __builtin_amdgcn_wmma_f32_16x16x32_bf16(
                    false, a_x[ks], false, bf, (short)0, acc, false, false);
            }

            // C/D layout: VGPR j -> M = j (lanes 0-15) / j+8 (lanes 16-31), N = col
#pragma unroll
            for (int j = 0; j < 8; ++j) {
                float y = acc[j] + bc[nt];
                y = (y >= 0.0f) ? y : al[nt] * y;
                op[j * HID + col] = y;
            }
        }
    }
}

extern "C" void kernel_launch(void* const* d_in, const int* in_sizes, int n_in,
                              void* d_out, int out_size, void* d_ws, size_t ws_size,
                              hipStream_t stream) {
    const float* x     = (const float*)d_in[0];
    const int*   ei    = (const int*)  d_in[1];
    const float* Wl    = (const float*)d_in[2];   // [4][64][64]
    const float* Wr    = (const float*)d_in[3];   // [4][64][64]
    const float* b     = (const float*)d_in[4];   // [4][64]
    const float* alpha = (const float*)d_in[5];   // [4][64]
    float* out = (float*)d_out;

    const int* src = ei;
    const int* dst = ei + N_EDGES;

    float* ws   = (float*)d_ws;
    float* deg  = ws;                                   // N floats (becomes deg_inv)
    float* agg  = deg  + N_NODES;                       // N*64
    float* buf0 = agg  + (size_t)N_NODES * HID;         // N*64
    float* buf1 = buf0 + (size_t)N_NODES * HID;         // N*64

    const long nfeat = (long)N_NODES * HID;

    // Degree -> deg_inv (once)
    zero_f32<<<(N_NODES + 255) / 256, 256, 0, stream>>>(deg, N_NODES);
    count_deg<<<(N_EDGES + 255) / 256, 256, 0, stream>>>(dst, deg);
    invert_deg<<<(N_NODES + 255) / 256, 256, 0, stream>>>(deg);

    const float* xc = x;
    float* outs[N_LAYERS] = { buf0, buf1, buf0, out };

    for (int l = 0; l < N_LAYERS; ++l) {
        zero_f32<<<(unsigned)((nfeat + 255) / 256), 256, 0, stream>>>(agg, nfeat);
        scatter_sum<<<(unsigned)(((long)N_EDGES * 16 + 255) / 256), 256, 0, stream>>>(
            xc, src, dst, agg);
        sage_fused_gemm<<<(N_NODES + 63) / 64, 128, 0, stream>>>(
            agg, xc,
            Wl + (size_t)l * HID * HID, Wr + (size_t)l * HID * HID,
            b + l * HID, alpha + l * HID,
            deg, outs[l]);
        xc = outs[l];
    }
}
